// CrossAttention_62457414419235
// MI455X (gfx1250) — matrix operand order repaired and checked by reading, not compiled, not run on hardware
//
#include <hip/hip_runtime.h>
#include <hip/hip_bf16.h>

typedef __attribute__((ext_vector_type(16))) __bf16 v16bf;
typedef __attribute__((ext_vector_type(8)))  __bf16 v8bf;
typedef __attribute__((ext_vector_type(8)))  float  v8f;

#define BB 2
#define HH 8
#define HD 32
#define NN 2048
#define DMODEL 256
#define SCALE_F 0.17677669529663687f   // 32^-0.5

__device__ inline v8f v8f_zero() {
  v8f z;
#pragma unroll
  for (int i = 0; i < 8; i++) z[i] = 0.f;
  return z;
}

__device__ inline v8f wmma_bf16(v16bf a, v16bf b, v8f c) {
  // (neg_a, A, neg_b, B, c_mod, C, reuse_a, reuse_b)
  return __builtin_amdgcn_wmma_f32_16x16x32_bf16(false, a, false, b, (short)0, c, false, false);
}

// A-fragment (16x32 bf16) loader: lane holds row = lane%16; element i -> K=hf*16+i.
// rowp points at the 32-element bf16 row.
__device__ inline v16bf load_a_bf16(const __bf16* rowp, int hf) {
  v8bf a0 = *(const v8bf*)(rowp + hf * 16);
  v8bf a1 = *(const v8bf*)(rowp + hf * 16 + 8);
  v16bf a;
#pragma unroll
  for (int i = 0; i < 8; i++) { a[i] = a0[i]; a[8 + i] = a1[i]; }
  return a;
}

// ---------------------------------------------------------------------------
// Weight convert + transpose: WT[n*K + k] = bf16(W[k*Ncol + n])
// ---------------------------------------------------------------------------
__global__ void transpose_bf16_kernel(const float* __restrict__ W,
                                      __bf16* __restrict__ WT, int K, int Ncol) {
  int idx = blockIdx.x * blockDim.x + threadIdx.x;
  if (idx >= K * Ncol) return;
  int k = idx / Ncol, n = idx % Ncol;
  WT[(size_t)n * K + k] = (__bf16)W[idx];
}

// ---------------------------------------------------------------------------
// Input projection: rv = X[BB*NN,256] @ W_in[256,512]; scatter to
// r[BB*HH][NN][32] and vT[BB*HH][32][NN] (v stored transposed), bf16.
// One wave per 16x16 output tile; K-loop of 8 x wmma_f32_16x16x32_bf16.
// ---------------------------------------------------------------------------
__global__ __launch_bounds__(256) void proj_in_kernel(
    const float* __restrict__ X, const __bf16* __restrict__ WT,
    __bf16* __restrict__ rOut, __bf16* __restrict__ vTOut) {
  int lane = threadIdx.x & 31, w = threadIdx.x >> 5;
  int hf = lane >> 4, l16 = lane & 15;
  int gwave = blockIdx.x * 8 + w;           // 8192 waves: 256 m-tiles x 32 n-tiles
  int m0 = (gwave >> 5) * 16;
  int n0 = (gwave & 31) * 16;

  v8f c = v8f_zero();
#pragma unroll
  for (int kb = 0; kb < DMODEL; kb += 32) {
    const float* xp = X + (size_t)(m0 + l16) * DMODEL + kb;
    v8f x0 = *(const v8f*)(xp + hf * 16);
    v8f x1 = *(const v8f*)(xp + hf * 16 + 8);
    v16bf a;
#pragma unroll
    for (int i = 0; i < 8; i++) { a[i] = (__bf16)x0[i]; a[8 + i] = (__bf16)x1[i]; }
    v16bf bfr = *(const v16bf*)(WT + (size_t)(n0 + l16) * DMODEL + kb + hf * 16);
    c = wmma_bf16(a, bfr, c);
  }

#pragma unroll
  for (int j = 0; j < 8; j++) {
    int M = m0 + j + hf * 8;                // global row in [0, BB*NN)
    int b = M >> 11;                        // / NN
    int n = M & (NN - 1);
    int col = n0 + l16;                     // in [0, 512)
    float val = c[j];
    if (col < DMODEL) {                     // r half
      int h = col >> 5, d = col & 31;
      rOut[(((size_t)b * HH + h) * NN + n) * HD + d] = (__bf16)val;
    } else {                                // v half, store transposed
      int cc = col - DMODEL;
      int h = cc >> 5, d = cc & 31;
      vTOut[(((size_t)b * HH + h) * HD + d) * NN + n] = (__bf16)val;
    }
  }
}

// ---------------------------------------------------------------------------
// Fused flash-style bidirectional attention (one direction per launch).
// S = (rA @ rB^T) * SCALE per (b,h). Wave owns a 16-row strip:
//   sweep 1: per-lane online (max,sum) over the columns each lane sees;
//            single cross-lane log-merge afterwards (exact softmax stats).
//   sweep 2: recompute S, P = exp(s-m)/sum -> bf16 via private LDS slice,
//            accumulate out[16,32] = P @ v via 2 WMMAs per 32-col chunk.
// Column-softmax direction = same kernel with rA/rB and v swapped.
// ---------------------------------------------------------------------------
__global__ __launch_bounds__(256) void attn_kernel(
    const __bf16* __restrict__ rA,   // [BB*HH][NN][32]
    const __bf16* __restrict__ rB,   // [BB*HH][NN][32]
    const __bf16* __restrict__ vT,   // [BB*HH][32][NN]
    __bf16* __restrict__ outF) {     // [BB*NN][256], cols h*32..h*32+31
  __shared__ __align__(64) __bf16 ldsP[8][16 * 32];
  int lane = threadIdx.x & 31, w = threadIdx.x >> 5;
  int hf = lane >> 4, l16 = lane & 15;
  int gwave = blockIdx.x * 8 + w;          // 2048 waves: (BB*HH) x 128 strips
  int bh = gwave >> 7;
  int s0 = (gwave & 127) * 16;
  const __bf16* rAp = rA + (size_t)bh * NN * HD;
  const __bf16* rBp = rB + (size_t)bh * NN * HD;
  const __bf16* vTp = vT + (size_t)bh * HD * NN;

  v16bf afr = load_a_bf16(rAp + (size_t)(s0 + l16) * HD, hf);

  // per-lane running (max, sum): lane covers cols {t0 + l16} of row (j + hf*8)
  float rmax[8], rsum[8];
#pragma unroll
  for (int j = 0; j < 8; j++) { rmax[j] = -3.0e38f; rsum[j] = 0.f; }

  // ---- sweep 1: per-lane online softmax stats (no in-loop shuffles) -------
  for (int t0 = 0; t0 < NN; t0 += 16) {
    int tp = (t0 + 64 < NN) ? (t0 + 64) : t0;
    __builtin_prefetch(rBp + (size_t)(tp + l16) * HD, 0, 0);
    v16bf bfr = *(const v16bf*)(rBp + (size_t)(t0 + l16) * HD + hf * 16);
    v8f c = wmma_bf16(afr, bfr, v8f_zero());
#pragma unroll
    for (int j = 0; j < 8; j++) {
      float v = c[j] * SCALE_F;
      float nm = fmaxf(rmax[j], v);
      rsum[j] = rsum[j] * __expf(rmax[j] - nm) + __expf(v - nm);
      rmax[j] = nm;
    }
  }
  // single cross-lane merge of (max,sum) pairs across the 16 lanes of the half
#pragma unroll
  for (int j = 0; j < 8; j++) {
#pragma unroll
    for (int m = 1; m < 16; m <<= 1) {
      float om = __shfl_xor(rmax[j], m, 32);
      float os = __shfl_xor(rsum[j], m, 32);
      float nm = fmaxf(rmax[j], om);
      rsum[j] = rsum[j] * __expf(rmax[j] - nm) + os * __expf(om - nm);
      rmax[j] = nm;
    }
  }
  float rinv[8];
#pragma unroll
  for (int j = 0; j < 8; j++) rinv[j] = 1.0f / rsum[j];

  // ---- sweep 2: recompute S, P @ v ----------------------------------------
  v8f acc0 = v8f_zero(), acc1 = v8f_zero();
  __bf16* Pw = &ldsP[w][0];
  for (int t0 = 0; t0 < NN; t0 += 32) {
    int tp = (t0 + 64 < NN) ? (t0 + 64) : t0;
    __builtin_prefetch(rBp + (size_t)(tp + l16) * HD, 0, 0);
    __builtin_prefetch(vTp + (size_t)l16 * NN + tp, 0, 0);
#pragma unroll
    for (int sub = 0; sub < 2; sub++) {
      v16bf bfr = *(const v16bf*)(rBp + (size_t)(t0 + sub * 16 + l16) * HD + hf * 16);
      v8f c = wmma_bf16(afr, bfr, v8f_zero());
#pragma unroll
      for (int j = 0; j < 8; j++) {
        float p = __expf(c[j] * SCALE_F - rmax[j]) * rinv[j];
        Pw[(j + hf * 8) * 32 + sub * 16 + l16] = (__bf16)p;  // [row][col]
      }
    }
    // same-wave DS ops are in-order: safe to re-fragment P as an A operand
    v16bf ap = load_a_bf16(Pw + l16 * 32, hf);
    v16bf bv0 = *(const v16bf*)(vTp + (size_t)(l16) * NN + t0 + hf * 16);
    v16bf bv1 = *(const v16bf*)(vTp + (size_t)(16 + l16) * NN + t0 + hf * 16);
    acc0 = wmma_bf16(ap, bv0, acc0);
    acc1 = wmma_bf16(ap, bv1, acc1);
  }

  int b = bh / HH, h = bh % HH;
#pragma unroll
  for (int j = 0; j < 8; j++) {
    int srow = s0 + j + hf * 8;
    size_t base = ((size_t)b * NN + srow) * DMODEL + h * HD;
    outF[base + l16]      = (__bf16)acc0[j];
    outF[base + 16 + l16] = (__bf16)acc1[j];
  }
}

// ---------------------------------------------------------------------------
// Output projection: Y = F[BB*NN,256] @ W_out[256,256] + bias, f32 out.
// ---------------------------------------------------------------------------
__global__ __launch_bounds__(256) void proj_out_kernel(
    const __bf16* __restrict__ F, const __bf16* __restrict__ WT,
    const float* __restrict__ bias, float* __restrict__ Y) {
  int lane = threadIdx.x & 31, w = threadIdx.x >> 5;
  int hf = lane >> 4, l16 = lane & 15;
  int gwave = blockIdx.x * 8 + w;           // 4096 waves: 256 m-tiles x 16 n-tiles
  int m0 = (gwave >> 4) * 16;
  int n0 = (gwave & 15) * 16;

  v8f c = v8f_zero();
#pragma unroll
  for (int kb = 0; kb < DMODEL; kb += 32) {
    v16bf a = load_a_bf16(F + (size_t)(m0 + l16) * DMODEL + kb, hf);
    v16bf bfr = *(const v16bf*)(WT + (size_t)(n0 + l16) * DMODEL + kb + hf * 16);
    c = wmma_bf16(a, bfr, c);
  }
  float bv = bias[n0 + l16];
#pragma unroll
  for (int j = 0; j < 8; j++) {
    int M = m0 + j + hf * 8;
    Y[(size_t)M * DMODEL + n0 + l16] = c[j] + bv;
  }
}

// ---------------------------------------------------------------------------
extern "C" void kernel_launch(void* const* d_in, const int* in_sizes, int n_in,
                              void* d_out, int out_size, void* d_ws, size_t ws_size,
                              hipStream_t stream) {
  const float* x_sed     = (const float*)d_in[0];
  const float* x_doa     = (const float*)d_in[1];
  const float* W_sed_in  = (const float*)d_in[2];
  const float* W_doa_in  = (const float*)d_in[3];
  const float* W_sed_out = (const float*)d_in[4];
  const float* b_sed_out = (const float*)d_in[5];
  const float* W_doa_out = (const float*)d_in[6];
  const float* b_doa_out = (const float*)d_in[7];
  float* out = (float*)d_out;

  __bf16* ws = (__bf16*)d_ws;
  size_t off = 0;
  __bf16* WT_sed_in  = ws + off; off += (size_t)512 * 256;
  __bf16* WT_doa_in  = ws + off; off += (size_t)512 * 256;
  __bf16* WT_sed_out = ws + off; off += (size_t)256 * 256;
  __bf16* WT_doa_out = ws + off; off += (size_t)256 * 256;
  __bf16* r_sed  = ws + off; off += (size_t)BB * HH * NN * HD;
  __bf16* r_doa  = ws + off; off += (size_t)BB * HH * NN * HD;
  __bf16* vT_sed = ws + off; off += (size_t)BB * HH * HD * NN;
  __bf16* vT_doa = ws + off; off += (size_t)BB * HH * HD * NN;
  __bf16* F_sed  = ws + off; off += (size_t)BB * NN * DMODEL;
  __bf16* F_doa  = ws + off; off += (size_t)BB * NN * DMODEL;

  // 1) weights -> transposed bf16
  transpose_bf16_kernel<<<(256 * 512 + 255) / 256, 256, 0, stream>>>(W_sed_in,  WT_sed_in,  256, 512);
  transpose_bf16_kernel<<<(256 * 512 + 255) / 256, 256, 0, stream>>>(W_doa_in,  WT_doa_in,  256, 512);
  transpose_bf16_kernel<<<(256 * 256 + 255) / 256, 256, 0, stream>>>(W_sed_out, WT_sed_out, 256, 256);
  transpose_bf16_kernel<<<(256 * 256 + 255) / 256, 256, 0, stream>>>(W_doa_out, WT_doa_out, 256, 256);

  // 2) input projections (r, v^T per head)
  proj_in_kernel<<<1024, 256, 0, stream>>>(x_sed, WT_sed_in, r_sed, vT_sed);
  proj_in_kernel<<<1024, 256, 0, stream>>>(x_doa, WT_doa_in, r_doa, vT_doa);

  // 3) bidirectional attention: row-softmax of S, and row-softmax of S^T
  attn_kernel<<<256, 256, 0, stream>>>(r_sed, r_doa, vT_doa, F_sed);
  attn_kernel<<<256, 256, 0, stream>>>(r_doa, r_sed, vT_sed, F_doa);

  // 4) output projections + bias
  proj_out_kernel<<<512, 256, 0, stream>>>(F_sed, WT_sed_out, b_sed_out, out);
  proj_out_kernel<<<512, 256, 0, stream>>>(F_doa, WT_doa_out, b_doa_out,
                                           out + (size_t)BB * NN * DMODEL);
}
